// BiLSTM_Enc_Dec_CRF_15753940042182
// MI455X (gfx1250) — matrix operand order repaired and checked by reading, not compiled, not run on hardware
//
#include <hip/hip_runtime.h>
#include <cstdint>
#include <cstddef>

// ---------------- problem constants ----------------
#define T_LEN 2048
#define HID   512
#define GDIM  2048          // 4*HID (gate dim)
#define DIN0  1088          // ELMO + POS
#define DIN1  1024          // 2*HID
#define KTAGS 48
#define NWG_CHAIN 8         // workgroups cooperating on one LSTM direction

typedef __bf16 bf16_t;
typedef __attribute__((ext_vector_type(16))) __bf16 v16bf;
typedef __attribute__((ext_vector_type(8)))  __bf16 v8bf;
typedef __attribute__((ext_vector_type(8)))  float  v8f;

__device__ __forceinline__ float sigmoidf_(float x) { return 1.0f / (1.0f + __expf(-x)); }

// ---------------- init: zero barrier counters + zero-state buffer ----------------
__global__ void init_ws(unsigned* __restrict__ counters, float* __restrict__ zeros) {
  int i = blockIdx.x * blockDim.x + threadIdx.x;
  if (i < 16) counters[i] = 0u;
  if (i < 2048) zeros[i] = 0.0f;
}

// ---------------- build embeds = concat(sentence, pos_emb[speech_tags]) as bf16 ----------------
__global__ void embed_kernel(const float* __restrict__ sent, const int* __restrict__ sp,
                             const float* __restrict__ pos, bf16_t* __restrict__ out) {
  int idx = blockIdx.x * blockDim.x + threadIdx.x;
  if (idx >= T_LEN * DIN0) return;
  int t = idx / DIN0, c = idx - t * DIN0;
  float v = (c < 1024) ? sent[t * 1024 + c] : pos[sp[t] * 64 + (c - 1024)];
  out[idx] = (bf16_t)v;
}

// ---------------- generic f32 -> bf16 conversion ----------------
__global__ void f32_to_bf16(const float* __restrict__ in, bf16_t* __restrict__ out, int n) {
  int idx = blockIdx.x * blockDim.x + threadIdx.x;
  if (idx < n) out[idx] = (bf16_t)in[idx];
}

// ---------------- WMMA bf16 GEMM: Out[d][t][n] = A[t] . W[d][n] + b_ih[d][n] + b_hh[d][n] ----------------
// A: (T_LEN x Kdim) bf16 row-major; W: (2 x GDIM x Kdim) bf16 row-major; Out: (2 x T_LEN x GDIM) f32
// Block = 256 thr (8 waves as 4x2); block tile 128(M) x 128(N); wave tile 32x64 (2x4 WMMA tiles).
__global__ __launch_bounds__(256)
void wmma_xp_gemm(const bf16_t* __restrict__ A, const bf16_t* __restrict__ W,
                  const float* __restrict__ b_ih, const float* __restrict__ b_hh,
                  float* __restrict__ Out, int Kdim) {
  const int N = GDIM;
  int d = blockIdx.z;
  const bf16_t* Wd = W + (size_t)d * N * Kdim;
  const float* bi = b_ih + (size_t)d * N;
  const float* bh = b_hh + (size_t)d * N;
  float* Od = Out + (size_t)d * T_LEN * N;

  int tid  = threadIdx.x;
  int lane = tid & 31;
  int wave = tid >> 5;
  int wm = wave & 3;                  // 4 M-bands of 32
  int wn = wave >> 2;                 // 2 N-bands of 64
  int m0 = blockIdx.y * 128 + wm * 32;
  int n0 = blockIdx.x * 128 + wn * 64;

  int m16 = lane & 15;
  int hiA = (lane >> 4) << 4;         // A frag K-base: 0 or 16
  int hiB = (lane >> 4) << 4;         // B frag K-base: 0 or 16

  v8f c00 = {}, c01 = {}, c02 = {}, c03 = {};
  v8f c10 = {}, c11 = {}, c12 = {}, c13 = {};

  const bf16_t* Arow0 = A  + (size_t)(m0 + m16) * Kdim;
  const bf16_t* Arow1 = Arow0 + (size_t)16 * Kdim;
  const bf16_t* Brow0 = Wd + (size_t)(n0 + m16) * Kdim;
  const bf16_t* Brow1 = Brow0 + (size_t)16 * Kdim;
  const bf16_t* Brow2 = Brow0 + (size_t)32 * Kdim;
  const bf16_t* Brow3 = Brow0 + (size_t)48 * Kdim;

  union AFrag { v16bf v; v8bf h[2]; };

  for (int k0 = 0; k0 < Kdim; k0 += 32) {
    AFrag a0, a1;
    a0.h[0] = *(const v8bf*)(Arow0 + k0 + hiA);
    a0.h[1] = *(const v8bf*)(Arow0 + k0 + hiA + 8);
    a1.h[0] = *(const v8bf*)(Arow1 + k0 + hiA);
    a1.h[1] = *(const v8bf*)(Arow1 + k0 + hiA + 8);
    v16bf b0 = *(const v16bf*)(Brow0 + k0 + hiB);
    v16bf b1 = *(const v16bf*)(Brow1 + k0 + hiB);
    v16bf b2 = *(const v16bf*)(Brow2 + k0 + hiB);
    v16bf b3 = *(const v16bf*)(Brow3 + k0 + hiB);
    if (k0 + 32 < Kdim) {             // pull next K-slab toward L1/L2
      __builtin_prefetch(Arow0 + k0 + 32, 0, 3);
      __builtin_prefetch(Brow0 + k0 + 32, 0, 3);
      __builtin_prefetch(Brow2 + k0 + 32, 0, 3);
    }
    c00 = __builtin_amdgcn_wmma_f32_16x16x32_bf16(false, a0.v, false, b0, (short)0, c00, false, false);
    c01 = __builtin_amdgcn_wmma_f32_16x16x32_bf16(false, a0.v, false, b1, (short)0, c01, false, false);
    c02 = __builtin_amdgcn_wmma_f32_16x16x32_bf16(false, a0.v, false, b2, (short)0, c02, false, false);
    c03 = __builtin_amdgcn_wmma_f32_16x16x32_bf16(false, a0.v, false, b3, (short)0, c03, false, false);
    c10 = __builtin_amdgcn_wmma_f32_16x16x32_bf16(false, a1.v, false, b0, (short)0, c10, false, false);
    c11 = __builtin_amdgcn_wmma_f32_16x16x32_bf16(false, a1.v, false, b1, (short)0, c11, false, false);
    c12 = __builtin_amdgcn_wmma_f32_16x16x32_bf16(false, a1.v, false, b2, (short)0, c12, false, false);
    c13 = __builtin_amdgcn_wmma_f32_16x16x32_bf16(false, a1.v, false, b3, (short)0, c13, false, false);
  }

  int rowOff = (lane >> 4) << 3;      // C frag: M = v (+8 for upper half-wave)
  int col    = lane & 15;
  auto store_tile = [&](const v8f& c, int mt, int nt) {
    int cc = nt + col;
    float bias = bi[cc] + bh[cc];
#pragma unroll
    for (int v = 0; v < 8; ++v)
      Od[(size_t)(mt + rowOff + v) * N + cc] = c[v] + bias;
  };
  store_tile(c00, m0,      n0);
  store_tile(c01, m0,      n0 + 16);
  store_tile(c02, m0,      n0 + 32);
  store_tile(c03, m0,      n0 + 48);
  store_tile(c10, m0 + 16, n0);
  store_tile(c11, m0 + 16, n0 + 16);
  store_tile(c12, m0 + 16, n0 + 32);
  store_tile(c13, m0 + 16, n0 + 48);
}

// ---------------- cooperative LSTM scan ----------------
// grid = (NWG_CHAIN, 2 dirs). Each WG keeps a 256-row bf16-packed slice of w_hh resident in LDS
// (256 x 512 x 2B at a 260-dword padded stride) for the whole 2048-step scan.
// ONE grid barrier per step: gate pre-activations -> barrier -> every WG redundantly computes the
// (identical) elementwise h/c update into its own LDS h; gates buffer is parity double-buffered,
// so a fast WG can never overwrite a buffer a slow WG is still reading.
__device__ __forceinline__ void grid_barrier(unsigned* cnt, unsigned target) {
  __threadfence();
  __syncthreads();
  if (threadIdx.x == 0) {
    atomicAdd(cnt, 1u);
    volatile unsigned* vc = (volatile unsigned*)cnt;
    while (*vc < target) __builtin_amdgcn_s_sleep(2);
  }
  __threadfence();
  __syncthreads();
}

#define LDSW_STRIDE 260   // pad: lane bank base = 4*lane mod 64 -> b128 reads spread banks
#define SCAN_SMEM_BYTES (256 * LDSW_STRIDE * 4 + 512 * 4)

__global__ __launch_bounds__(256)
void lstm_scan(const float* __restrict__ xp,     // (2, T, GDIM) incl. biases
               const float* __restrict__ w_hh,   // (2, GDIM, HID) f32
               const float* __restrict__ h0,     // (2, HID) for this layer
               const float* __restrict__ c0,     // (2, HID)
               float* __restrict__ out,          // (T, 2*HID)  [fwd | bwd]
               float* __restrict__ hT,           // (2, HID)
               float* __restrict__ cT,           // (2, HID)
               float* __restrict__ gates,        // (2 dirs, 2 parity, GDIM)
               unsigned* __restrict__ counters)  // [2]
{
  extern __shared__ unsigned char smem[];
  unsigned* lds_w = (unsigned*)smem;                        // [row][LDSW_STRIDE] packed bf16 pairs
  float*    lds_h = (float*)(smem + 256 * LDSW_STRIDE * 4); // [512]

  int d   = blockIdx.y;
  int wg  = blockIdx.x;
  int tid = threadIdx.x;
  int r0  = wg * 256;                       // this WG's gate rows [r0, r0+256)
  const float* whd = w_hh + (size_t)d * GDIM * HID;

  // Preload + pack: thread tid owns row r0+tid; dword i holds bf16(w[2i]),bf16(w[2i+1]).
  {
    const float4* gp = (const float4*)(whd + (size_t)(r0 + tid) * HID);
    unsigned* wrow = lds_w + (size_t)tid * LDSW_STRIDE;
    for (int q = 0; q < 64; ++q) {
      float4 a = gp[2 * q], b = gp[2 * q + 1];
      uint4 u;
      u.x = ((__float_as_uint(a.x) + 0x8000u) >> 16) | ((__float_as_uint(a.y) + 0x8000u) & 0xffff0000u);
      u.y = ((__float_as_uint(a.z) + 0x8000u) >> 16) | ((__float_as_uint(a.w) + 0x8000u) & 0xffff0000u);
      u.z = ((__float_as_uint(b.x) + 0x8000u) >> 16) | ((__float_as_uint(b.y) + 0x8000u) & 0xffff0000u);
      u.w = ((__float_as_uint(b.z) + 0x8000u) >> 16) | ((__float_as_uint(b.w) + 0x8000u) & 0xffff0000u);
      *(uint4*)(wrow + 4 * q) = u;
    }
  }
  // initial h into LDS; c replicated into registers of every WG
  lds_h[tid]       = h0[d * HID + tid];
  lds_h[tid + 256] = h0[d * HID + tid + 256];
  float creg0 = c0[d * HID + tid];
  float creg1 = c0[d * HID + tid + 256];
  float hreg0 = 0.f, hreg1 = 0.f;
  __syncthreads();

  unsigned bar = 0;
  unsigned* cnt = counters + d;
  bool rev = (d == 1);

  for (int s = 0; s < T_LEN; ++s) {
    int t = rev ? (T_LEN - 1 - s) : s;

    // gate pre-activation for row r0+tid: dot(w_row, h) via b128 LDS reads
    float acc = 0.0f;
    const unsigned* wrow = lds_w + (size_t)tid * LDSW_STRIDE;
#pragma unroll 4
    for (int i = 0; i < 64; ++i) {
      uint4 u = *(const uint4*)(wrow + 4 * i);
      const float4* hp = (const float4*)(lds_h + 8 * i);
      float4 hA = hp[0], hB = hp[1];
      acc = fmaf(__uint_as_float(u.x << 16),         hA.x, acc);
      acc = fmaf(__uint_as_float(u.x & 0xffff0000u), hA.y, acc);
      acc = fmaf(__uint_as_float(u.y << 16),         hA.z, acc);
      acc = fmaf(__uint_as_float(u.y & 0xffff0000u), hA.w, acc);
      acc = fmaf(__uint_as_float(u.z << 16),         hB.x, acc);
      acc = fmaf(__uint_as_float(u.z & 0xffff0000u), hB.y, acc);
      acc = fmaf(__uint_as_float(u.w << 16),         hB.z, acc);
      acc = fmaf(__uint_as_float(u.w & 0xffff0000u), hB.w, acc);
    }
    int row = r0 + tid;
    float* gbuf = gates + (size_t)(d * 2 + (s & 1)) * GDIM;
    gbuf[row] = acc + xp[((size_t)d * T_LEN + t) * GDIM + row];

    bar += NWG_CHAIN;
    grid_barrier(cnt, bar);                  // all gates of this step visible

    // replicated elementwise update (i,f,g,o gate order); identical in every WG
#pragma unroll
    for (int e = 0; e < 2; ++e) {
      int j = tid + e * 256;
      float ig = gbuf[j];
      float fg = gbuf[HID + j];
      float gg = gbuf[2 * HID + j];
      float og = gbuf[3 * HID + j];
      float cprev = e ? creg1 : creg0;
      float cnew = sigmoidf_(fg) * cprev + sigmoidf_(ig) * tanhf(gg);
      float hh = sigmoidf_(og) * tanhf(cnew);
      if (e) { creg1 = cnew; hreg1 = hh; } else { creg0 = cnew; hreg0 = hh; }
      lds_h[j] = hh;
      if (wg == 0) out[(size_t)t * (2 * HID) + d * HID + j] = hh;
    }
    __syncthreads();                          // lds_h ready for next step's dot
  }

  if (wg == 0) {
    hT[d * HID + tid] = hreg0; hT[d * HID + tid + 256] = hreg1;
    cT[d * HID + tid] = creg0; cT[d * HID + tid + 256] = creg1;
  }
}

// ---------------- init-state projections: [init_h|init_c] = W @ enc_state + b ----------------
__global__ void proj_init(const float* __restrict__ enc_h, const float* __restrict__ enc_c,
                          const float* __restrict__ e2h_w, const float* __restrict__ e2h_b,
                          const float* __restrict__ e2c_w, const float* __restrict__ e2c_b,
                          float* __restrict__ init_h, float* __restrict__ init_c) {
  int o = blockIdx.x * blockDim.x + threadIdx.x;   // 0..4095
  int sel = o >> 11, j = o & 2047;
  const float* W = sel ? e2c_w : e2h_w;
  const float* x = sel ? enc_c : enc_h;
  const float* b = sel ? e2c_b : e2h_b;
  const float* wr = W + (size_t)j * 2048;
  float acc = b[j];
  for (int k = 0; k < 2048; ++k) acc = fmaf(wr[k], x[k], acc);
  (sel ? init_c : init_h)[j] = acc;
}

// ---------------- feats = dec_out @ h2t_w.T + h2t_b ----------------
__global__ void feats_kernel(const float* __restrict__ dec_out, const float* __restrict__ h2t_w,
                             const float* __restrict__ h2t_b, float* __restrict__ feats) {
  int idx = blockIdx.x * blockDim.x + threadIdx.x;
  if (idx >= T_LEN * KTAGS) return;
  int t = idx / KTAGS, k = idx - t * KTAGS;
  const float* x = dec_out + (size_t)t * 1024;
  const float* w = h2t_w + (size_t)k * 1024;
  float acc = h2t_b[k];
  for (int n = 0; n < 1024; ++n) acc = fmaf(x[n], w[n], acc);
  feats[idx] = acc;
}

// ---------------- CRF: forward partition - gold score (START=0, END=1) ----------------
__global__ void crf_kernel(const float* __restrict__ feats, const int* __restrict__ tags,
                           const float* __restrict__ trans, float* __restrict__ out) {
  __shared__ float tr[KTAGS * KTAGS];
  __shared__ float fv[KTAGS];
  __shared__ float nfv[KTAGS];
  __shared__ float red[64];
  __shared__ float fwd_s;
  int tid = threadIdx.x;
  for (int i = tid; i < KTAGS * KTAGS; i += 64) tr[i] = trans[i];
  if (tid < KTAGS) fv[tid] = (tid == 0) ? 0.0f : -10000.0f;
  __syncthreads();

  for (int t = 0; t < T_LEN; ++t) {
    if (tid < KTAGS) {
      const float* trow = tr + tid * KTAGS;
      float m = -3.0e38f;
      for (int p = 0; p < KTAGS; ++p) m = fmaxf(m, fv[p] + trow[p]);
      float s = 0.0f;
      for (int p = 0; p < KTAGS; ++p) s += __expf(fv[p] + trow[p] - m);
      nfv[tid] = feats[t * KTAGS + tid] + m + __logf(s);
    }
    __syncthreads();
    if (tid < KTAGS) fv[tid] = nfv[tid];
    __syncthreads();
  }
  if (tid == 0) {
    const float* trow = tr + 1 * KTAGS;        // END row
    float m = -3.0e38f;
    for (int p = 0; p < KTAGS; ++p) m = fmaxf(m, fv[p] + trow[p]);
    float s = 0.0f;
    for (int p = 0; p < KTAGS; ++p) s += __expf(fv[p] + trow[p] - m);
    fwd_s = m + __logf(s);
  }
  float loc = 0.0f;
  for (int t = tid; t < T_LEN; t += 64) {
    int cur = tags[t];
    int prev = (t == 0) ? 0 : tags[t - 1];
    loc += tr[cur * KTAGS + prev] + feats[t * KTAGS + cur];
  }
  red[tid] = loc;
  __syncthreads();
  for (int s = 32; s > 0; s >>= 1) { if (tid < s) red[tid] += red[tid + s]; __syncthreads(); }
  if (tid == 0) {
    float gold = red[0] + tr[1 * KTAGS + tags[T_LEN - 1]];
    out[0] = fwd_s - gold;
  }
}

// ---------------- host: orchestrate ----------------
extern "C" void kernel_launch(void* const* d_in, const int* in_sizes, int n_in,
                              void* d_out, int out_size, void* d_ws, size_t ws_size,
                              hipStream_t stream) {
  (void)in_sizes; (void)n_in; (void)out_size; (void)ws_size;
  const float* sentence    = (const float*)d_in[0];
  const int*   speech_tags = (const int*)d_in[1];
  const int*   tags        = (const int*)d_in[2];
  const float* pos_emb     = (const float*)d_in[3];
  const float* enc_w_ih0 = (const float*)d_in[4];
  const float* enc_w_hh0 = (const float*)d_in[5];
  const float* enc_b_ih0 = (const float*)d_in[6];
  const float* enc_b_hh0 = (const float*)d_in[7];
  const float* enc_w_ih1 = (const float*)d_in[8];
  const float* enc_w_hh1 = (const float*)d_in[9];
  const float* enc_b_ih1 = (const float*)d_in[10];
  const float* enc_b_hh1 = (const float*)d_in[11];
  const float* dec_w_ih0 = (const float*)d_in[12];
  const float* dec_w_hh0 = (const float*)d_in[13];
  const float* dec_b_ih0 = (const float*)d_in[14];
  const float* dec_b_hh0 = (const float*)d_in[15];
  const float* dec_w_ih1 = (const float*)d_in[16];
  const float* dec_w_hh1 = (const float*)d_in[17];
  const float* dec_b_ih1 = (const float*)d_in[18];
  const float* dec_b_hh1 = (const float*)d_in[19];
  const float* h2t_w = (const float*)d_in[20];
  const float* h2t_b = (const float*)d_in[21];
  const float* e2h_w = (const float*)d_in[22];
  const float* e2h_b = (const float*)d_in[23];
  const float* e2c_w = (const float*)d_in[24];
  const float* e2c_b = (const float*)d_in[25];
  const float* trans = (const float*)d_in[26];

  char* ws = (char*)d_ws;
  size_t off = 0;
  auto take = [&](size_t bytes) -> char* {
    char* p = ws + off;
    off = (off + bytes + 255) & ~(size_t)255;
    return p;
  };
  bf16_t*   EMB   = (bf16_t*)take((size_t)T_LEN * DIN0 * 2);
  bf16_t*   SEQ   = (bf16_t*)take((size_t)T_LEN * DIN1 * 2);
  bf16_t*   WB    = (bf16_t*)take((size_t)2 * GDIM * DIN0 * 2);
  float*    XP    = (float*)take((size_t)2 * T_LEN * GDIM * 4);
  float*    OUT0  = (float*)take((size_t)T_LEN * 1024 * 4);
  float*    OUT1  = (float*)take((size_t)T_LEN * 1024 * 4);
  float*    ENCH  = (float*)take(2048 * 4);
  float*    ENCC  = (float*)take(2048 * 4);
  float*    INH   = (float*)take(2048 * 4);
  float*    INC   = (float*)take(2048 * 4);
  float*    GTS   = (float*)take((size_t)4 * GDIM * 4);   // 2 dirs x 2 parity x GDIM
  float*    FEATS = (float*)take((size_t)T_LEN * KTAGS * 4);
  float*    ZER   = (float*)take(2048 * 4);
  unsigned* CNT   = (unsigned*)take(64);

  dim3 gemm_grid(GDIM / 128, T_LEN / 128, 2);
  dim3 scan_grid(NWG_CHAIN, 2);
  const size_t SMEM = SCAN_SMEM_BYTES;

  init_ws<<<8, 256, 0, stream>>>(CNT, ZER);
  embed_kernel<<<(T_LEN * DIN0 + 255) / 256, 256, 0, stream>>>(sentence, speech_tags, pos_emb, EMB);

  // ---- encoder layer 0 ----
  f32_to_bf16<<<(2 * GDIM * DIN0 + 255) / 256, 256, 0, stream>>>(enc_w_ih0, WB, 2 * GDIM * DIN0);
  wmma_xp_gemm<<<gemm_grid, 256, 0, stream>>>(EMB, WB, enc_b_ih0, enc_b_hh0, XP, DIN0);
  lstm_scan<<<scan_grid, 256, SMEM, stream>>>(XP, enc_w_hh0, ZER, ZER, OUT0, ENCH, ENCC, GTS, CNT + 0);

  // ---- encoder layer 1 ----
  f32_to_bf16<<<(T_LEN * 1024 + 255) / 256, 256, 0, stream>>>(OUT0, SEQ, T_LEN * 1024);
  f32_to_bf16<<<(2 * GDIM * DIN1 + 255) / 256, 256, 0, stream>>>(enc_w_ih1, WB, 2 * GDIM * DIN1);
  wmma_xp_gemm<<<gemm_grid, 256, 0, stream>>>(SEQ, WB, enc_b_ih1, enc_b_hh1, XP, DIN1);
  lstm_scan<<<scan_grid, 256, SMEM, stream>>>(XP, enc_w_hh1, ZER, ZER, OUT1, ENCH + 1024, ENCC + 1024, GTS, CNT + 2);

  // ---- decoder init states ----
  proj_init<<<16, 256, 0, stream>>>(ENCH, ENCC, e2h_w, e2h_b, e2c_w, e2c_b, INH, INC);

  // ---- decoder layer 0 ----
  f32_to_bf16<<<(2 * GDIM * DIN0 + 255) / 256, 256, 0, stream>>>(dec_w_ih0, WB, 2 * GDIM * DIN0);
  wmma_xp_gemm<<<gemm_grid, 256, 0, stream>>>(EMB, WB, dec_b_ih0, dec_b_hh0, XP, DIN0);
  lstm_scan<<<scan_grid, 256, SMEM, stream>>>(XP, dec_w_hh0, INH, INC, OUT0, ENCH, ENCC /*dump*/, GTS, CNT + 4);

  // ---- decoder layer 1 ----
  f32_to_bf16<<<(T_LEN * 1024 + 255) / 256, 256, 0, stream>>>(OUT0, SEQ, T_LEN * 1024);
  f32_to_bf16<<<(2 * GDIM * DIN1 + 255) / 256, 256, 0, stream>>>(dec_w_ih1, WB, 2 * GDIM * DIN1);
  wmma_xp_gemm<<<gemm_grid, 256, 0, stream>>>(SEQ, WB, dec_b_ih1, dec_b_hh1, XP, DIN1);
  lstm_scan<<<scan_grid, 256, SMEM, stream>>>(XP, dec_w_hh1, INH + 1024, INC + 1024, OUT1, ENCH, ENCC /*dump*/, GTS, CNT + 6);

  // ---- CRF head ----
  feats_kernel<<<(T_LEN * KTAGS + 255) / 256, 256, 0, stream>>>(OUT1, h2t_w, h2t_b, FEATS);
  crf_kernel<<<1, 64, 0, stream>>>(FEATS, tags, trans, (float*)d_out);
}
